// LSTMModule_87162066305493
// MI455X (gfx1250) — matrix operand order repaired and checked by reading, not compiled, not run on hardware
//
#include <hip/hip_runtime.h>

// ---------------------------------------------------------------------------
// Bidirectional LSTM (B=64, T=2048, IN=256, H=128) + mixer, for gfx1250.
// Phase 1: xp GEMM (bf16 WMMA, f32 acc)  -- fully parallel
// Phase 2: recurrence, 8 independent WGs (2 dirs x 4 batch groups), w_hh in
//          VGPRs, h/c/gates in LDS, xp double-buffered in registers
// Phase 3: mixer GEMM (bf16 WMMA, f32 acc) -> d_out (f32)
// ---------------------------------------------------------------------------

typedef __bf16 bf16;
typedef __attribute__((ext_vector_type(16))) __bf16 v16bf;
typedef __attribute__((ext_vector_type(8)))  __bf16 v8bf;
typedef __attribute__((ext_vector_type(8)))  float  v8f;

#define B_   64
#define T_   2048
#define IN_  256
#define H_   128
#define G_   512            // 4*H
#define OUT_ 256
#define BT_  (B_*T_)

// ---- workspace layout (bytes). Requires ~641 MiB of d_ws. ----------------
static constexpr size_t OFF_XBF  = 0;                                   // BT*IN bf16   = 64 MiB
static constexpr size_t OFF_WIH  = OFF_XBF  + (size_t)BT_*IN_*2;        // 2*G*IN bf16  = 512 KiB
static constexpr size_t OFF_WHH  = OFF_WIH  + (size_t)2*G_*IN_*2;       // 2*G*H  bf16  = 256 KiB
static constexpr size_t OFF_MW   = OFF_WHH  + (size_t)2*G_*H_*2;        // OUT*2H bf16  = 128 KiB
static constexpr size_t OFF_BIAS = OFF_MW   + (size_t)OUT_*2*H_*2;      // 2*G f32      = 4 KiB
static constexpr size_t OFF_XP   = OFF_BIAS + (size_t)2*G_*4;           // 2*T*B*G f32  = 512 MiB (time-major)
static constexpr size_t OFF_HS   = OFF_XP   + (size_t)2*BT_*G_*4;       // B*T*2H bf16  = 64 MiB

__device__ inline v8f v8f_splat(float x) {
  v8f r;
#pragma unroll
  for (int i = 0; i < 8; ++i) r[i] = x;
  return r;
}

__device__ inline v16bf cat8(v8bf lo, v8bf hi) {
  return __builtin_shufflevector(lo, hi, 0,1,2,3,4,5,6,7,8,9,10,11,12,13,14,15);
}

__device__ inline v8f wmma_bf16(v16bf a, v16bf b, v8f c) {
  // D(16x16,f32) = A(16x32,bf16) x B(32x16,bf16) + C
  return __builtin_amdgcn_wmma_f32_16x16x32_bf16(
      /*neg_a=*/false, a, /*neg_b=*/false, b,
      /*c_mod=*/(short)0, c, /*reuse_a=*/false, /*reuse_b=*/false);
}

__device__ inline float sigmoidf_(float x) { return 1.f / (1.f + __expf(-x)); }

// ---------------------------------------------------------------------------
__global__ void cvt_f32_to_bf16(bf16* __restrict__ dst, const float* __restrict__ src, int n) {
  int i = blockIdx.x * blockDim.x + threadIdx.x;
  if (i < n) dst[i] = (bf16)src[i];
}

__global__ void bias_combine(float* __restrict__ dst, const float* __restrict__ bi,
                             const float* __restrict__ bh, int n) {
  int i = blockIdx.x * blockDim.x + threadIdx.x;
  if (i < n) dst[i] = bi[i] + bh[i];
}

// ---------------------------------------------------------------------------
// Phase 1: xp[dir][t][b][g] = sum_k x[b,t,k]*w_ih[dir][g,k] + (b_ih+b_hh)[g]
// One wave = one M-tile (16 consecutive t of one b) x 4 N-tiles (64 gates).
// grid = (BT/16/8, G/64, 2), block = 256 (8 waves).
// ---------------------------------------------------------------------------
__global__ __launch_bounds__(256) void xp_gemm(
    const bf16* __restrict__ xbf, const bf16* __restrict__ wih,
    const float* __restrict__ biasc, float* __restrict__ xp)
{
  const int wv = threadIdx.x >> 5, ln = threadIdx.x & 31;
  const int lh = ln >> 4, lm = ln & 15;
  const int r0 = (blockIdx.x * 8 + wv) * 16;   // flat row in (B,T)
  const int b  = r0 >> 11;                     // r0 / T
  const int t0 = r0 & (T_ - 1);
  const int n0 = blockIdx.y * 64;
  const int dir = blockIdx.z;

  const bf16*  W  = wih   + (size_t)dir * G_ * IN_;
  const float* bs = biasc + dir * G_;
  float*       XP = xp    + (size_t)dir * (size_t)BT_ * G_;

  v8f acc[4];
#pragma unroll
  for (int i = 0; i < 4; ++i) acc[i] = v8f_splat(bs[n0 + 16*i + lm]);

#pragma unroll
  for (int kk = 0; kk < 8; ++kk) {             // K = 256 = 8 x 32
    // A fragment: lane (lh,lm) holds row m=lm, K = kk*32 + lh*8 + {0..7, 16..23}
    const bf16* ap = xbf + (size_t)(r0 + lm) * IN_ + kk*32 + lh*8;
    v16bf a = cat8(*(const v8bf*)ap, *(const v8bf*)(ap + 16));
#pragma unroll
    for (int i = 0; i < 4; ++i) {
      // B fragment: lane holds col n=n0+16i+lm, K = kk*32 + lh*8 + {0..7, 16..23}
      const bf16* bp = W + (size_t)(n0 + 16*i + lm) * IN_ + kk*32 + lh*8;
      acc[i] = wmma_bf16(a, cat8(*(const v8bf*)bp, *(const v8bf*)(bp + 16)), acc[i]);
    }
  }
  // store: C layout -> element (m = j + 8*lh, n), xp is (T,B,G)
#pragma unroll
  for (int i = 0; i < 4; ++i)
#pragma unroll
    for (int j = 0; j < 8; ++j)
      XP[((size_t)(t0 + j + 8*lh) * B_ + b) * G_ + n0 + 16*i + lm] = acc[i][j];
}

// ---------------------------------------------------------------------------
// Phase 2: recurrence. grid = (4 batch groups, 2 dirs), block = 1024 (32 waves).
// Wave w owns gate columns [16w, 16w+16) for all 16 batch rows of this WG.
// w_hh fragments persist in VGPRs; per step: 4 WMMAs + LDS gates + cell update.
// ---------------------------------------------------------------------------
__global__ __launch_bounds__(1024) void lstm_rec(
    const float* __restrict__ xp, const bf16* __restrict__ whh,
    bf16* __restrict__ hs)
{
  __shared__ float gates[16][G_];   // 32 KB
  __shared__ float cbuf[16][H_];    //  8 KB
  __shared__ bf16  hbuf[16][H_];    //  4 KB

  const int dir = blockIdx.y;
  const int b0  = blockIdx.x * 16;
  const int wv = threadIdx.x >> 5, ln = threadIdx.x & 31;
  const int lh = ln >> 4, lm = ln & 15;
  const int n0 = wv * 16;

  const float* XP = xp  + (size_t)dir * (size_t)BT_ * G_;
  const bf16*  W  = whh + (size_t)dir * G_ * H_;

  // persistent B fragments: w_hh^T slice, K=128 in 4 chunks of 32
  v16bf bfr[4];
#pragma unroll
  for (int kk = 0; kk < 4; ++kk)
    bfr[kk] = cat8(*(const v8bf*)(W + (size_t)(n0 + lm) * H_ + kk*32 + lh*8),
                   *(const v8bf*)(W + (size_t)(n0 + lm) * H_ + kk*32 + lh*8 + 16));

  for (int i = threadIdx.x; i < 16 * H_; i += 1024) {
    ((float*)cbuf)[i] = 0.f;
    ((bf16*)hbuf)[i]  = (bf16)0.f;
  }
  __syncthreads();

  const int jj = threadIdx.x & (H_ - 1);
  const int m0 = threadIdx.x >> 7;

  // xp tile for the first step (acts as WMMA accumulator init: gates = xp + h*Whh^T)
  v8f cur;
  {
    const int t = dir ? (T_ - 1) : 0;
#pragma unroll
    for (int j = 0; j < 8; ++j)
      cur[j] = XP[((size_t)t * B_ + b0 + j + 8*lh) * G_ + n0 + lm];
  }

  for (int it = 0; it < T_; ++it) {
    const int t = dir ? (T_ - 1 - it) : it;

    // prefetch next step's xp tile into registers (hidden behind this step)
    v8f nxt = v8f_splat(0.f);
    if (it + 1 < T_) {
      const int tn = dir ? (t - 1) : (t + 1);
#pragma unroll
      for (int j = 0; j < 8; ++j)
        nxt[j] = XP[((size_t)tn * B_ + b0 + j + 8*lh) * G_ + n0 + lm];
    }

    // gates tile = xp_t + h @ w_hh^T  (4 chained WMMAs, K=128)
    v8f acc = cur;
#pragma unroll
    for (int kk = 0; kk < 4; ++kk) {
      const bf16* ap = &hbuf[lm][kk*32 + lh*8];
      v16bf a = cat8(*(const v8bf*)ap, *(const v8bf*)(ap + 16));
      acc = wmma_bf16(a, bfr[kk], acc);
    }
#pragma unroll
    for (int j = 0; j < 8; ++j)
      gates[j + 8*lh][n0 + lm] = acc[j];
    __syncthreads();

    // LSTM cell: 16x128 elems over 1024 threads (2 each)
#pragma unroll
    for (int m = m0; m < 16; m += 8) {
      float iv = gates[m][jj];
      float fv = gates[m][H_   + jj];
      float gv = gates[m][2*H_ + jj];
      float ov = gates[m][3*H_ + jj];
      float c  = sigmoidf_(fv) * cbuf[m][jj] + sigmoidf_(iv) * tanhf(gv);
      float h  = sigmoidf_(ov) * tanhf(c);
      cbuf[m][jj] = c;
      hbuf[m][jj] = (bf16)h;
      hs[((size_t)(b0 + m) * T_ + t) * (2*H_) + dir*H_ + jj] = (bf16)h;
    }
    __syncthreads();   // hbuf/gates safe for next iteration
    cur = nxt;
  }
}

// ---------------------------------------------------------------------------
// Phase 3: out = hs @ mixer_w^T + mixer_b.  grid = (BT/16/8, OUT/64), block 256.
// ---------------------------------------------------------------------------
__global__ __launch_bounds__(256) void mixer_gemm(
    const bf16* __restrict__ hs, const bf16* __restrict__ mw,
    const float* __restrict__ mb, float* __restrict__ out)
{
  const int wv = threadIdx.x >> 5, ln = threadIdx.x & 31;
  const int lh = ln >> 4, lm = ln & 15;
  const int r0 = (blockIdx.x * 8 + wv) * 16;
  const int n0 = blockIdx.y * 64;

  v8f acc[4];
#pragma unroll
  for (int i = 0; i < 4; ++i) acc[i] = v8f_splat(mb[n0 + 16*i + lm]);

#pragma unroll
  for (int kk = 0; kk < 8; ++kk) {             // K = 256
    const bf16* ap = hs + (size_t)(r0 + lm) * (2*H_) + kk*32 + lh*8;
    v16bf a = cat8(*(const v8bf*)ap, *(const v8bf*)(ap + 16));
#pragma unroll
    for (int i = 0; i < 4; ++i) {
      const bf16* bp = mw + (size_t)(n0 + 16*i + lm) * (2*H_) + kk*32 + lh*8;
      acc[i] = wmma_bf16(a, cat8(*(const v8bf*)bp, *(const v8bf*)(bp + 16)), acc[i]);
    }
  }
#pragma unroll
  for (int i = 0; i < 4; ++i)
#pragma unroll
    for (int j = 0; j < 8; ++j)
      out[(size_t)(r0 + j + 8*lh) * OUT_ + n0 + 16*i + lm] = acc[i][j];
}

// ---------------------------------------------------------------------------
extern "C" void kernel_launch(void* const* d_in, const int* in_sizes, int n_in,
                              void* d_out, int out_size, void* d_ws, size_t ws_size,
                              hipStream_t stream) {
  (void)in_sizes; (void)n_in; (void)out_size; (void)ws_size;
  const float* x     = (const float*)d_in[0];
  const float* wih_f = (const float*)d_in[1];
  const float* whh_f = (const float*)d_in[2];
  const float* bih_f = (const float*)d_in[3];
  const float* bhh_f = (const float*)d_in[4];
  const float* wih_b = (const float*)d_in[5];
  const float* whh_b = (const float*)d_in[6];
  const float* bih_b = (const float*)d_in[7];
  const float* bhh_b = (const float*)d_in[8];
  const float* mw    = (const float*)d_in[9];
  const float* mb    = (const float*)d_in[10];
  float* out = (float*)d_out;

  char* ws = (char*)d_ws;
  bf16*  xbf   = (bf16*) (ws + OFF_XBF);
  bf16*  wihbf = (bf16*) (ws + OFF_WIH);
  bf16*  whhbf = (bf16*) (ws + OFF_WHH);
  bf16*  mwbf  = (bf16*) (ws + OFF_MW);
  float* biasc = (float*)(ws + OFF_BIAS);
  float* xp    = (float*)(ws + OFF_XP);
  bf16*  hs    = (bf16*) (ws + OFF_HS);

  // precision packing
  cvt_f32_to_bf16<<<(BT_*IN_ + 255)/256, 256, 0, stream>>>(xbf, x, BT_*IN_);
  cvt_f32_to_bf16<<<(G_*IN_  + 255)/256, 256, 0, stream>>>(wihbf,            wih_f, G_*IN_);
  cvt_f32_to_bf16<<<(G_*IN_  + 255)/256, 256, 0, stream>>>(wihbf + G_*IN_,   wih_b, G_*IN_);
  cvt_f32_to_bf16<<<(G_*H_   + 255)/256, 256, 0, stream>>>(whhbf,            whh_f, G_*H_);
  cvt_f32_to_bf16<<<(G_*H_   + 255)/256, 256, 0, stream>>>(whhbf + G_*H_,    whh_b, G_*H_);
  cvt_f32_to_bf16<<<(OUT_*2*H_ + 255)/256, 256, 0, stream>>>(mwbf, mw, OUT_*2*H_);
  bias_combine<<<(G_ + 255)/256, 256, 0, stream>>>(biasc,       bih_f, bhh_f, G_);
  bias_combine<<<(G_ + 255)/256, 256, 0, stream>>>(biasc + G_,  bih_b, bhh_b, G_);

  // phase 1: input projection for both directions
  xp_gemm<<<dim3(BT_/16/8, G_/64, 2), 256, 0, stream>>>(xbf, wihbf, biasc, xp);
  // phase 2: recurrence, 8 independent workgroups
  lstm_rec<<<dim3(B_/16, 2), 1024, 0, stream>>>(xp, whhbf, hs);
  // phase 3: mixer
  mixer_gemm<<<dim3(BT_/16/8, OUT_/64), 256, 0, stream>>>(hs, mwbf, mb, out);
}
